// DictSeparateActorResnBinary_37993280700485
// MI455X (gfx1250) — compile-verified, matrix operand order checked by reading, not run on hardware
//
#include <hip/hip_runtime.h>
#include <hip/hip_bf16.h>

typedef __attribute__((ext_vector_type(16))) _Float16 v16h;
typedef __attribute__((ext_vector_type(8)))  float    v8f;

#define B_      128
#define C_      128
#define H_      24
#define W_      24
#define M_      16
#define NPATCH  15
#define P_      225
#define P_PAD   240
#define DEMB_   32
#define FIN_    265            // C + 137
#define K_PAD   288            // FIN padded to multiple of 32
#define A_STR   296            // LDS row stride in halves (bank-conflict pad)
#define UNITS   (B_ * M_)      // 2048
#define BLK     512            // 16 waves

// LDS layout (dynamic)
#define OFF_W    0                       // 128*296 fp16  = 75776 B
#define OFF_A    75776                   // 240*296 fp16  = 142080 B
#define OFF_M    217856                  // 240*128 fp16  = 61440 B
#define OFF_SAP  279296                  // 240 f32       = 960 B
#define OFF_RED  280256                  // 512 f32       = 2048 B
#define OFF_BIAS 282304                  // 128 f32
#define OFF_SAPW 282816                  // 128 f32
#define SMEM_SZ  283328

#define NEG_INF (-__builtin_inff())

// ---------------------------------------------------------------------------
// Prep: merger_w (f32, [128][265]) -> padded fp16 [128][A_STR] in workspace.
// ---------------------------------------------------------------------------
__global__ __launch_bounds__(256)
void convert_w_kernel(const float* __restrict__ merger_w,
                      _Float16* __restrict__ Wg)
{
    int idx = blockIdx.x * 256 + threadIdx.x;
    if (idx >= C_ * A_STR) return;
    int c = idx / A_STR, f = idx % A_STR;
    float v = (f < FIN_) ? merger_w[c * FIN_ + f] : 0.f;
    Wg[idx] = (_Float16)v;
}

// ---------------------------------------------------------------------------
// Fused per-unit kernel: async-stage W -> LDS, gather/build A tile -> LDS,
// WMMA GEMM (LDS-resident), leaky-relu, sap head, move head, masks + argmax.
// One 512-thread workgroup per (b, unit); ~283 KB LDS -> one WGP each.
// ---------------------------------------------------------------------------
__global__ __launch_bounds__(BLK)
void fused_actor_kernel(const float* __restrict__ x,
                        const float* __restrict__ pred,
                        const int* __restrict__ xcord,
                        const int* __restrict__ ycord,
                        const int* __restrict__ embf,
                        const float* __restrict__ contf,
                        const float* __restrict__ emb_table,
                        const _Float16* __restrict__ Wg,
                        const float* __restrict__ merger_b,
                        const float* __restrict__ sap_w,
                        const float* __restrict__ sap_b,
                        const float* __restrict__ move_w,
                        const float* __restrict__ move_b,
                        const unsigned char* __restrict__ aamw,
                        const unsigned char* __restrict__ aams,
                        float* __restrict__ out_wl,
                        float* __restrict__ out_sl,
                        float* __restrict__ out_wa,
                        float* __restrict__ out_sa)
{
    extern __shared__ char smem[];
    _Float16* Wl   = (_Float16*)(smem + OFF_W);    // [128][A_STR]
    _Float16* Al   = (_Float16*)(smem + OFF_A);    // [240][A_STR]
    _Float16* Ml   = (_Float16*)(smem + OFF_M);    // [240][128] merged
    float*    sapl = (float*)(smem + OFF_SAP);     // [240]
    float*    red  = (float*)(smem + OFF_RED);     // [512]
    float*    bias = (float*)(smem + OFF_BIAS);    // [128]
    float*    sapw = (float*)(smem + OFF_SAPW);    // [128]

    __shared__ float unitf[133];
    __shared__ int   pix[P_], piy[P_];
    __shared__ float mv6[6];
    __shared__ int   sbidx[BLK];

    const int tid  = threadIdx.x;
    const int unit = blockIdx.x;
    const int b    = unit >> 4;

    // ---- async-stage fp16 weights into LDS (ASYNCcnt-tracked DMA path) ----
    {
        unsigned lds0 = (unsigned)(size_t)(void*)Wl;
        unsigned long long g0 = (unsigned long long)(size_t)(const void*)Wg;
        for (int off = tid * 16; off < C_ * A_STR * 2; off += BLK * 16) {
            unsigned la = lds0 + (unsigned)off;
            unsigned long long ga = g0 + (unsigned long long)off;
            asm volatile("global_load_async_to_lds_b128 %0, %1, off"
                         :: "v"(la), "v"(ga) : "memory");
        }
    }
    if (tid < C_) { bias[tid] = merger_b[tid]; sapw[tid] = sap_w[tid]; }

    // ---- per-unit scalars (embeddings, continuous, patch indices) ----
    if (tid < 128) {                                   // 4 x 32 embedding dims
        int e = embf[unit * 4 + (tid >> 5)];
        unitf[tid] = emb_table[e * DEMB_ + (tid & 31)];
    } else if (tid < 133) {
        unitf[tid] = contf[unit * 5 + (tid - 128)];
    }
    if (tid < P_) {
        int i = tid / NPATCH, j = tid % NPATCH;
        pix[tid] = ycord[unit] + j - 7;                // width  (swapped coords)
        piy[tid] = xcord[unit] + i - 7;                // height
    }
    __syncthreads();

    // ---- gather/build fp16 A tile in LDS (grid_sample == integer offsets) ---
    for (int idx = tid; idx < P_PAD * K_PAD; idx += BLK) {
        int p = idx / K_PAD, f = idx % K_PAD;
        float v = 0.f;
        if (p < P_) {
            int ix = pix[p], iy = piy[p];
            bool valid = (ix >= 0) & (ix < W_) & (iy >= 0) & (iy < H_);
            int cx = ix < 0 ? 0 : (ix > W_ - 1 ? W_ - 1 : ix);
            int cy = iy < 0 ? 0 : (iy > H_ - 1 ? H_ - 1 : iy);
            if (f < C_) {
                float g = x[(((size_t)b * C_ + f) * H_ + cy) * W_ + cx];
                v = valid ? g : 0.f;
            } else if (f < C_ + 3) {
                float g = pred[(((size_t)b * 3 + (f - C_)) * H_ + cy) * W_ + cx];
                v = valid ? g : 0.f;
            } else if (f == C_ + 3) {
                v = valid ? 1.f : 0.f;                 // in_bounds channel
            } else if (f < FIN_) {
                v = unitf[f - (C_ + 4)];               // per-unit broadcast
            }
        }
        Al[p * A_STR + f] = (_Float16)v;
    }

    asm volatile("s_wait_asynccnt 0x0" ::: "memory");  // W DMA complete
    __syncthreads();

    // ---- LDS-resident WMMA GEMM: waves 0..14 each own a 16x128 tile --------
    const int wave = tid >> 5;
    const int lane = tid & 31;
    if (wave < 15) {
        const int row0  = wave * 16;
        const int mrow  = lane & 15;
        const int khalf = (lane < 16) ? 0 : 8;   // ISA 16-bit A/B K split

        v8f acc[8] = {};
        const _Float16* Arow = Al + (row0 + mrow) * A_STR;

        for (int kc = 0; kc < 9; ++kc) {
            const int k0 = kc * 32 + khalf;
            union { v16h h; uint4 q[2]; } af;
            af.q[0] = *(const uint4*)(Arow + k0);
            af.q[1] = *(const uint4*)(Arow + k0 + 16);
            union { v16h h; uint4 q[2]; } bf[8];
            #pragma unroll
            for (int nb = 0; nb < 8; ++nb) {
                const int n = nb * 16 + (lane & 15);
                bf[nb].q[0] = *(const uint4*)(Wl + n * A_STR + k0);
                bf[nb].q[1] = *(const uint4*)(Wl + n * A_STR + k0 + 16);
            }
            #pragma unroll
            for (int nb = 0; nb < 8; ++nb)
                acc[nb] = __builtin_amdgcn_wmma_f32_16x16x32_f16(
                    false, af.h, false, bf[nb].h, (short)0, acc[nb], false, false);
        }

        // Epilogue: bias + leaky_relu -> merged LDS (fp16), fused sap dot.
        const float sb = sap_b[0];
        float psum[8];
        #pragma unroll
        for (int r = 0; r < 8; ++r) psum[r] = 0.f;
        #pragma unroll
        for (int nb = 0; nb < 8; ++nb) {
            const int col = nb * 16 + (lane & 15);
            const float bv = bias[col];
            const float sw = sapw[col];
            #pragma unroll
            for (int r = 0; r < 8; ++r) {
                float v   = acc[nb][r] + bv;
                float act = v > 0.f ? v : 0.01f * v;
                const int m = r + ((lane >= 16) ? 8 : 0);   // C/D layout
                Ml[(row0 + m) * C_ + col] = (_Float16)act;
                psum[r] += act * sw;
            }
        }
        #pragma unroll
        for (int r = 0; r < 8; ++r) {
            float s = psum[r];
            s += __shfl_xor(s, 1, 32);
            s += __shfl_xor(s, 2, 32);
            s += __shfl_xor(s, 4, 32);
            s += __shfl_xor(s, 8, 32);   // reduce N within each 16-lane half
            if ((lane & 15) == 0) {
                const int m = r + ((lane >= 16) ? 8 : 0);
                sapl[row0 + m] = s + sb;
            }
        }
    }
    __syncthreads();

    // ---- move head: 6 dots of 28800 over LDS-resident merged ---------------
    {
        float acc6[6] = {0.f, 0.f, 0.f, 0.f, 0.f, 0.f};
        for (int idx = tid; idx < P_ * C_; idx += BLK) {
            int p = idx >> 7, c = idx & 127;
            float v = (float)Ml[p * C_ + c];
            #pragma unroll
            for (int a = 0; a < 6; ++a)
                acc6[a] += v * move_w[(size_t)a * (P_ * C_) + idx];
        }
        for (int a = 0; a < 6; ++a) {
            red[tid] = acc6[a];
            __syncthreads();
            for (int s = BLK / 2; s > 0; s >>= 1) {
                if (tid < s) red[tid] += red[tid + s];
                __syncthreads();
            }
            if (tid == 0) mv6[a] = red[0] + move_b[a];
            __syncthreads();
        }
    }

    // ---- worker head: mask + logits + first-max argmax ---------------------
    if (tid == 0) {
        bool m[6]; bool any = false;
        #pragma unroll
        for (int a = 0; a < 6; ++a) { m[a] = aamw[unit * 6 + a] != 0; any = any || m[a]; }
        float best = NEG_INF; int bi = 0;
        #pragma unroll
        for (int a = 0; a < 6; ++a) {
            float v = (any && !m[a]) ? NEG_INF : mv6[a];
            out_wl[unit * 6 + a] = v;
            if (v > best) { best = v; bi = a; }
        }
        out_wa[unit] = (float)bi;
    }

    // ---- sapper head: mask + logits + first-max argmax ---------------------
    {
        const bool in = tid < P_;
        const bool mv = in ? (aams[(size_t)unit * P_ + tid] != 0) : false;
        red[tid] = mv ? 1.f : 0.f;
        __syncthreads();
        for (int s = BLK / 2; s > 0; s >>= 1) {
            if (tid < s) red[tid] += red[tid + s];
            __syncthreads();
        }
        const bool any = red[0] > 0.f;
        __syncthreads();
        float v = NEG_INF;
        if (in) {
            v = sapl[tid];
            if (any && !mv) v = NEG_INF;
            out_sl[(size_t)unit * P_ + tid] = v;
        }
        red[tid]   = v;
        sbidx[tid] = in ? tid : 0x7fffffff;
        __syncthreads();
        for (int s = BLK / 2; s > 0; s >>= 1) {
            if (tid < s) {
                float ov = red[tid + s]; int oi = sbidx[tid + s];
                if (ov > red[tid] || (ov == red[tid] && oi < sbidx[tid])) {
                    red[tid] = ov; sbidx[tid] = oi;
                }
            }
            __syncthreads();
        }
        if (tid == 0) out_sa[unit] = (float)sbidx[0];
    }
}

// ---------------------------------------------------------------------------
extern "C" void kernel_launch(void* const* d_in, const int* in_sizes, int n_in,
                              void* d_out, int out_size, void* d_ws, size_t ws_size,
                              hipStream_t stream)
{
    const float* x         = (const float*)d_in[0];
    const float* pred      = (const float*)d_in[1];
    const int*   xcord     = (const int*)d_in[2];
    const int*   ycord     = (const int*)d_in[3];
    const int*   embf      = (const int*)d_in[4];
    const float* contf     = (const float*)d_in[5];
    const unsigned char* aamw = (const unsigned char*)d_in[6];
    const unsigned char* aams = (const unsigned char*)d_in[7];
    const float* emb_table = (const float*)d_in[8];
    const float* merger_w  = (const float*)d_in[9];
    const float* merger_b  = (const float*)d_in[10];
    const float* sap_w     = (const float*)d_in[11];
    const float* sap_b     = (const float*)d_in[12];
    const float* move_w    = (const float*)d_in[13];
    const float* move_b    = (const float*)d_in[14];

    _Float16* Wg = (_Float16*)d_ws;                 // padded fp16 weights

    float* out    = (float*)d_out;
    float* out_wl = out;                            // 2048*6
    float* out_sl = out + UNITS * 6;                // 2048*225
    float* out_wa = out + UNITS * 6 + UNITS * P_;   // 2048 (as float)
    float* out_sa = out_wa + UNITS;                 // 2048 (as float)

    convert_w_kernel<<<(C_ * A_STR + 255) / 256, 256, 0, stream>>>(merger_w, Wg);

    fused_actor_kernel<<<UNITS, BLK, SMEM_SZ, stream>>>(
        x, pred, xcord, ycord, embf, contf, emb_table, Wg,
        merger_b, sap_w, sap_b, move_w, move_b, aamw, aams,
        out_wl, out_sl, out_wa, out_sa);
}
